// RWKV7_CrossAttention_78391743086653
// MI455X (gfx1250) — matrix operand order repaired and checked by reading, not compiled, not run on hardware
//
#include <hip/hip_runtime.h>
#include <hip/hip_bf16.h>
#include <math.h>

// ---------------- problem constants (from reference) ----------------
#define Hc    256
#define Tc    512
#define EBc   16
#define NHc   4
#define HDc   64
#define MROWS (EBc*Tc)                   // 8192
#define NEGc  (-0.6065306597126334f)     // -exp(-0.5)
#define EPSc  (6.4e-4f)                  // HD * 1e-5

typedef __attribute__((ext_vector_type(16))) __bf16 v16bf;
typedef __attribute__((ext_vector_type(8)))  __bf16 v8bf;
typedef __attribute__((ext_vector_type(8)))  float  v8f;

static __device__ __forceinline__ float sigm(float x) { return 1.f / (1.f + expf(-x)); }

// per-head (64-thread group) reduction inside a 256-thread block
static __device__ __forceinline__ float reduce64(float* red, int c) {
  const int j = c & 63;
  __syncthreads();
#pragma unroll
  for (int s = 32; s >= 1; s >>= 1) {
    if (j < s) red[c] += red[c + s];
    __syncthreads();
  }
  float r = red[c - j];
  __syncthreads();
  return r;
}

// ---------------- weight packing ----------------
__global__ void cvt_bf16_kernel(const float* __restrict__ src, __bf16* __restrict__ dst, int n) {
  int i = blockIdx.x * 256 + threadIdx.x;
  if (i < n) dst[i] = (__bf16)src[i];
}
__global__ void transpose_bf16_kernel(const float* __restrict__ src, __bf16* __restrict__ dst,
                                      int R, int C) {
  int i = blockIdx.x * 256 + threadIdx.x;
  if (i >= R * C) return;
  int r = i / C, c = i - r * C;
  dst[(size_t)c * R + r] = (__bf16)src[i];
}

// ---------------- token mixing (bf16 activations) ----------------
__global__ void __launch_bounds__(256)
mix_kernel(const float* __restrict__ query, const float* __restrict__ keyval,
           const float* __restrict__ x_r, const float* __restrict__ x_w,
           const float* __restrict__ x_k, const float* __restrict__ x_v,
           const float* __restrict__ x_a, const float* __restrict__ x_g,
           __bf16* __restrict__ Xw, __bf16* __restrict__ Xk,
           __bf16* __restrict__ Xv, __bf16* __restrict__ Xa,
           __bf16* __restrict__ XrL, __bf16* __restrict__ XgL) {
  const int idx = blockIdx.x * 256 + threadIdx.x;      // over MROWS*Hc, exact
  const int c   = idx & (Hc - 1);
  const int row = idx >> 8;                            // eb*Tc + t
  const int t   = row & (Tc - 1);
  const int eb  = row >> 9;
  const int b   = eb >> 3;
  const float hs = keyval[((size_t)b * Tc + t) * Hc + c];
  const float eq = query[(size_t)eb * Hc + c];
  const float pv = (t > 0) ? keyval[((size_t)b * Tc + t - 1) * Hc + c] : 0.f;
  const float xc  = eq - hs;
  const float xkv = pv - hs;
  Xw[idx] = (__bf16)(hs + xc  * x_w[c]);
  Xa[idx] = (__bf16)(hs + xc  * x_a[c]);
  Xk[idx] = (__bf16)(hs + xkv * x_k[c]);
  Xv[idx] = (__bf16)(hs + xkv * x_v[c]);
  if (t == Tc - 1) {                                    // only last step feeds r / g / epilogue
    XrL[eb * Hc + c] = (__bf16)(hs + xc * x_r[c]);
    XgL[eb * Hc + c] = (__bf16)(hs + xc * x_g[c]);
  }
}

// ---------------- generic WMMA GEMM: C[MxN] f32 = A[MxK] * B[NxK]^T, bf16 in ----------------
// One wave per 16x16 output tile, K in steps of 32 (v_wmma_f32_16x16x32_bf16).
__global__ void __launch_bounds__(128)
gemm_bf16_wmma(const __bf16* __restrict__ A, const __bf16* __restrict__ B,
               float* __restrict__ C, int M, int N, int K) {
  const int lane = threadIdx.x & 31;
  const int wave = blockIdx.x * (blockDim.x >> 5) + (threadIdx.x >> 5);
  const int tnc  = N >> 4;
  const int total = (M >> 4) * tnc;
  if (wave >= total) return;                           // wave-uniform: EXEC stays all-ones
  const int tm = wave / tnc;
  const int tn = wave - tm * tnc;
  const int lr = lane & 15;
  const int hi = lane >> 4;
  // A fragment: lane row = lr, K chunks [hi*8, +8) and [hi*8+16, +8)  (ISA 16-bit A layout)
  const __bf16* Ap = A + (size_t)(tm * 16 + lr) * K + hi * 8;
  // B fragment: lane col = lr, K chunks [hi*8, +8) and [hi*8+16, +8)  (matches A)
  const __bf16* Bp = B + (size_t)(tn * 16 + lr) * K + hi * 8;
  v8f acc = {};
  for (int k0 = 0; k0 < K; k0 += 32) {
    v8bf a_lo = *(const v8bf*)(Ap + k0);
    v8bf a_hi = *(const v8bf*)(Ap + k0 + 16);
    v8bf b_lo = *(const v8bf*)(Bp + k0);
    v8bf b_hi = *(const v8bf*)(Bp + k0 + 16);
    v16bf av, bv;
#pragma unroll
    for (int i = 0; i < 8; ++i) {
      av[i] = a_lo[i]; av[i + 8] = a_hi[i];
      bv[i] = b_lo[i]; bv[i + 8] = b_hi[i];
    }
    acc = __builtin_amdgcn_wmma_f32_16x16x32_bf16(false, av, false, bv,
                                                  (short)0, acc, false, false);
  }
  // D layout: VGPR i -> row tm*16 + i + hi*8, col tn*16 + lr
  float* Cp = C + (size_t)(tm * 16 + hi * 8) * N + tn * 16 + lr;
#pragma unroll
  for (int i = 0; i < 8; ++i) Cp[(size_t)i * N] = acc[i];
}

// ---------------- elementwise glue ----------------
__global__ void ew_w_kernel(float* __restrict__ EW, const float* __restrict__ w0, int n) {
  int i = blockIdx.x * 256 + threadIdx.x;
  if (i >= n) return;
  int c = i & (Hc - 1);
  EW[i] = expf(NEGc * sigm(w0[c] + EW[i]));            // exp(w_t), pre-baked for the scan
}
__global__ void a_sig_kernel(float* __restrict__ AM, const float* __restrict__ a0, int n) {
  int i = blockIdx.x * 256 + threadIdx.x;
  if (i >= n) return;
  int c = i & (Hc - 1);
  AM[i] = sigm(a0[c] + AM[i]);
}
__global__ void v_update_kernel(float* __restrict__ VM, const float* __restrict__ VG,
                                const float* __restrict__ v0, const float* __restrict__ vfirst,
                                int n) {
  int i = blockIdx.x * 256 + threadIdx.x;
  if (i >= n) return;
  int c = i & (Hc - 1);
  float s = sigm(v0[c] + VG[i]);
  float v = VM[i];
  VM[i] = v + (vfirst[i] - v) * s;
}
__global__ void tanh_cvt_kernel(const float* __restrict__ src, __bf16* __restrict__ dst, int n) {
  int i = blockIdx.x * 256 + threadIdx.x;
  if (i < n) dst[i] = (__bf16)tanhf(src[i]);
}
__global__ void cvt_f2b_kernel(const float* __restrict__ src, __bf16* __restrict__ dst, int n) {
  int i = blockIdx.x * 256 + threadIdx.x;
  if (i < n) dst[i] = (__bf16)src[i];
}
__global__ void sig_cvt_kernel(const float* __restrict__ src, __bf16* __restrict__ dst, int n) {
  int i = blockIdx.x * 256 + threadIdx.x;
  if (i < n) dst[i] = (__bf16)sigm(src[i]);
}

// kk normalize + k_final + scan coefficient tensors (-kk, kk*a). One block per row.
__global__ void __launch_bounds__(256)
kk_kernel(float* __restrict__ KM, const float* __restrict__ AM,
          const float* __restrict__ k_k, const float* __restrict__ k_a,
          float* __restrict__ NKK, float* __restrict__ BM) {
  __shared__ float red[Hc];
  const size_t row = blockIdx.x;
  const int c = threadIdx.x;
  const size_t idx = row * Hc + c;
  const float kv  = KM[idx];
  const float av  = AM[idx];
  const float kkv = kv * k_k[c];
  red[c] = kkv * kkv;
  const float ss  = reduce64(red, c);
  const float inv = 1.f / fmaxf(sqrtf(ss), 1e-12f);
  const float kkn = kkv * inv;
  NKK[idx] = -kkn;
  BM[idx]  = kkn * av;
  KM[idx]  = kv + kv * (av - 1.f) * k_a[c];            // in-place k_final
}

// ---------------- sequential RWKV7 scan: one wave per (seq, head) ----------------
// State S[64 rows k][64 cols v] lives in 128 VGPRs/lane (lane owns cols lane, lane+32).
// Broadcast coefficients are wave-uniform -> scalar (SMEM) loads feeding VALU FMAs.
#define SCAN_WPB 4
__global__ void __launch_bounds__(128)
scan_kernel(const float* __restrict__ EW, const float* __restrict__ KM,
            const float* __restrict__ VM, const float* __restrict__ NKK,
            const float* __restrict__ BM, const float* __restrict__ RL,
            float* __restrict__ OS) {
  const int lane = threadIdx.x & 31;
  int unit = blockIdx.x * SCAN_WPB + (threadIdx.x >> 5);     // 0..63
  unit = __builtin_amdgcn_readfirstlane(unit);               // prove wave-uniformity
  const int eb = unit >> 2;
  const int h  = unit & 3;
  const size_t rowbase = (size_t)eb * Tc * Hc + h * HDc;
  float S[128];
#pragma unroll
  for (int i = 0; i < 128; ++i) S[i] = 0.f;
  for (int t = 0; t < Tc; ++t) {
    const size_t off = rowbase + (size_t)t * Hc;
    const float* __restrict__ ew = EW  + off;
    const float* __restrict__ km = KM  + off;
    const float* __restrict__ nk = NKK + off;
    const float* __restrict__ bm = BM  + off;
    const float v0 = VM[off + lane];
    const float v1 = VM[off + lane + 32];
    float as0 = 0.f, as1 = 0.f;
#pragma unroll
    for (int k = 0; k < HDc; ++k) {
      const float a = nk[k];
      as0 += a * S[2 * k];
      as1 += a * S[2 * k + 1];
    }
#pragma unroll
    for (int k = 0; k < HDc; ++k) {
      const float e  = ew[k];
      const float kc = km[k];
      const float bb = bm[k];
      S[2 * k]     = S[2 * k]     * e + kc * v0 + bb * as0;
      S[2 * k + 1] = S[2 * k + 1] * e + kc * v1 + bb * as1;
    }
  }
  // readout only needed at the last step: o = r_{T-1} . S
  float o0 = 0.f, o1 = 0.f;
  const float* __restrict__ r = RL + eb * Hc + h * HDc;
#pragma unroll
  for (int k = 0; k < HDc; ++k) {
    const float rv = r[k];
    o0 += rv * S[2 * k];
    o1 += rv * S[2 * k + 1];
  }
  OS[eb * Hc + h * HDc + lane]      = o0;
  OS[eb * Hc + h * HDc + lane + 32] = o1;
}

// ---------------- last-step epilogue: groupnorm + r_k residual + gate ----------------
__global__ void __launch_bounds__(256)
epilogue_kernel(const float* __restrict__ OS, const float* __restrict__ RL,
                const float* __restrict__ KM, const float* __restrict__ VM,
                const float* __restrict__ GL, const float* __restrict__ gn_w,
                const float* __restrict__ gn_b, const float* __restrict__ r_k,
                __bf16* __restrict__ OG) {
  __shared__ float red[Hc];
  const int eb = blockIdx.x;
  const int c  = threadIdx.x;
  const float o = OS[eb * Hc + c];
  red[c] = o;     const float s1 = reduce64(red, c);
  red[c] = o * o; const float s2 = reduce64(red, c);
  const float mu  = s1 * (1.f / HDc);
  const float var = s2 * (1.f / HDc) - mu * mu;
  float y = (o - mu) * rsqrtf(var + EPSc) * gn_w[c] + gn_b[c];
  const size_t last = ((size_t)eb * Tc + (Tc - 1)) * Hc + c;
  const float rl = RL[eb * Hc + c];
  const float kl = KM[last];
  red[c] = rl * kl * r_k[c];
  const float dot = reduce64(red, c);
  y += dot * VM[last];
  y *= GL[eb * Hc + c];
  OG[eb * Hc + c] = (__bf16)y;
}

// ------------------------------------------------------------------
extern "C" void kernel_launch(void* const* d_in, const int* in_sizes, int n_in,
                              void* d_out, int out_size, void* d_ws, size_t ws_size,
                              hipStream_t stream) {
  const float* query  = (const float*)d_in[0];
  const float* keyval = (const float*)d_in[1];
  const float* vfirst = (const float*)d_in[2];
  const float* x_r = (const float*)d_in[3];
  const float* x_w = (const float*)d_in[4];
  const float* x_k = (const float*)d_in[5];
  const float* x_v = (const float*)d_in[6];
  const float* x_a = (const float*)d_in[7];
  const float* x_g = (const float*)d_in[8];
  const float* w0 = (const float*)d_in[9];
  const float* w1 = (const float*)d_in[10];
  const float* w2 = (const float*)d_in[11];
  const float* a0 = (const float*)d_in[12];
  const float* a1 = (const float*)d_in[13];
  const float* a2 = (const float*)d_in[14];
  const float* v0 = (const float*)d_in[15];
  const float* v1 = (const float*)d_in[16];
  const float* v2 = (const float*)d_in[17];
  const float* g1 = (const float*)d_in[18];
  const float* g2 = (const float*)d_in[19];
  const float* k_k = (const float*)d_in[20];
  const float* k_a = (const float*)d_in[21];
  const float* r_k = (const float*)d_in[22];
  const float* Wr = (const float*)d_in[23];
  const float* Wk = (const float*)d_in[24];
  const float* Wv = (const float*)d_in[25];
  const float* Wo = (const float*)d_in[26];
  const float* gn_w = (const float*)d_in[27];
  const float* gn_b = (const float*)d_in[28];
  float* out = (float*)d_out;

  // ---- workspace bump allocator (~78 MB total) ----
  char* p = (char*)d_ws;
  auto alloc = [&](size_t bytes) -> char* {
    char* r = p;
    p += (bytes + 255) & ~(size_t)255;
    return r;
  };
  const size_t NBIG = (size_t)MROWS * Hc;        // 8192*256
  const size_t NSM  = (size_t)MROWS * 32;        // 8192*32
  __bf16* WrB = (__bf16*)alloc(Hc * Hc * 2);
  __bf16* WkB = (__bf16*)alloc(Hc * Hc * 2);
  __bf16* WvB = (__bf16*)alloc(Hc * Hc * 2);
  __bf16* WoB = (__bf16*)alloc(Hc * Hc * 2);
  __bf16* w1T = (__bf16*)alloc(32 * Hc * 2);
  __bf16* a1T = (__bf16*)alloc(32 * Hc * 2);
  __bf16* v1T = (__bf16*)alloc(32 * Hc * 2);
  __bf16* w2T = (__bf16*)alloc(Hc * 32 * 2);
  __bf16* a2T = (__bf16*)alloc(Hc * 32 * 2);
  __bf16* v2T = (__bf16*)alloc(Hc * 32 * 2);
  __bf16* g1T = (__bf16*)alloc(64 * Hc * 2);
  __bf16* g2T = (__bf16*)alloc(Hc * 64 * 2);
  __bf16* Xw = (__bf16*)alloc(NBIG * 2);
  __bf16* Xk = (__bf16*)alloc(NBIG * 2);
  __bf16* Xv = (__bf16*)alloc(NBIG * 2);
  __bf16* Xa = (__bf16*)alloc(NBIG * 2);
  __bf16* XrL = (__bf16*)alloc(EBc * Hc * 2);
  __bf16* XgL = (__bf16*)alloc(EBc * Hc * 2);
  float* KM  = (float*)alloc(NBIG * 4);
  float* VM  = (float*)alloc(NBIG * 4);
  float* EW  = (float*)alloc(NBIG * 4);
  float* AM  = (float*)alloc(NBIG * 4);
  float* NKK = (float*)alloc(NBIG * 4);
  float* BM  = (float*)alloc(NBIG * 4);
  float* VG  = (float*)alloc(NBIG * 4);
  float* HW  = (float*)alloc(NSM * 4);
  float* HA  = (float*)alloc(NSM * 4);
  float* HV  = (float*)alloc(NSM * 4);
  __bf16* TW = (__bf16*)alloc(NSM * 2);
  __bf16* TA = (__bf16*)alloc(NSM * 2);
  __bf16* TV = (__bf16*)alloc(NSM * 2);
  float* RL  = (float*)alloc(EBc * Hc * 4);
  float* HG  = (float*)alloc(EBc * 64 * 4);
  __bf16* TG = (__bf16*)alloc(EBc * 64 * 2);
  float* GL  = (float*)alloc(EBc * Hc * 4);
  float* OS  = (float*)alloc(EBc * Hc * 4);
  __bf16* OG = (__bf16*)alloc(EBc * Hc * 2);

  auto blks = [](size_t n) { return (int)((n + 255) / 256); };
  auto gemm = [&](const __bf16* A, const __bf16* B, float* C, int M, int N, int K) {
    int tiles = (M / 16) * (N / 16);
    gemm_bf16_wmma<<<(tiles + 3) / 4, 128, 0, stream>>>(A, B, C, M, N, K);
  };

  // 1) pack weights to bf16 (transposed where GEMM needs A@W form as A@B^T)
  cvt_bf16_kernel<<<blks(Hc * Hc), 256, 0, stream>>>(Wr, WrB, Hc * Hc);
  cvt_bf16_kernel<<<blks(Hc * Hc), 256, 0, stream>>>(Wk, WkB, Hc * Hc);
  cvt_bf16_kernel<<<blks(Hc * Hc), 256, 0, stream>>>(Wv, WvB, Hc * Hc);
  cvt_bf16_kernel<<<blks(Hc * Hc), 256, 0, stream>>>(Wo, WoB, Hc * Hc);
  transpose_bf16_kernel<<<blks(Hc * 32), 256, 0, stream>>>(w1, w1T, Hc, 32);
  transpose_bf16_kernel<<<blks(Hc * 32), 256, 0, stream>>>(a1, a1T, Hc, 32);
  transpose_bf16_kernel<<<blks(Hc * 32), 256, 0, stream>>>(v1, v1T, Hc, 32);
  transpose_bf16_kernel<<<blks(32 * Hc), 256, 0, stream>>>(w2, w2T, 32, Hc);
  transpose_bf16_kernel<<<blks(32 * Hc), 256, 0, stream>>>(a2, a2T, 32, Hc);
  transpose_bf16_kernel<<<blks(32 * Hc), 256, 0, stream>>>(v2, v2T, 32, Hc);
  transpose_bf16_kernel<<<blks(Hc * 64), 256, 0, stream>>>(g1, g1T, Hc, 64);
  transpose_bf16_kernel<<<blks(64 * Hc), 256, 0, stream>>>(g2, g2T, 64, Hc);

  // 2) token mixing
  mix_kernel<<<(int)(NBIG / 256), 256, 0, stream>>>(query, keyval, x_r, x_w, x_k, x_v, x_a, x_g,
                                                    Xw, Xk, Xv, Xa, XrL, XgL);

  // 3) big + low-rank projections (WMMA)
  gemm(Xk, WkB, KM, MROWS, Hc, Hc);     // k_raw
  gemm(Xv, WvB, VM, MROWS, Hc, Hc);     // v_raw
  gemm(Xw, w1T, HW, MROWS, 32, Hc);
  gemm(Xa, a1T, HA, MROWS, 32, Hc);
  gemm(Xv, v1T, HV, MROWS, 32, Hc);
  tanh_cvt_kernel<<<blks(NSM), 256, 0, stream>>>(HW, TW, (int)NSM);
  cvt_f2b_kernel<<<blks(NSM), 256, 0, stream>>>(HA, TA, (int)NSM);
  cvt_f2b_kernel<<<blks(NSM), 256, 0, stream>>>(HV, TV, (int)NSM);
  gemm(TW, w2T, EW, MROWS, Hc, 32);
  gemm(TA, a2T, AM, MROWS, Hc, 32);
  gemm(TV, v2T, VG, MROWS, Hc, 32);

  // 4) nonlinear glue
  ew_w_kernel<<<blks(NBIG), 256, 0, stream>>>(EW, w0, (int)NBIG);          // exp(w)
  a_sig_kernel<<<blks(NBIG), 256, 0, stream>>>(AM, a0, (int)NBIG);         // a gate
  v_update_kernel<<<blks(NBIG), 256, 0, stream>>>(VM, VG, v0, vfirst, (int)NBIG);
  kk_kernel<<<MROWS, 256, 0, stream>>>(KM, AM, k_k, k_a, NKK, BM);         // -kk, kk*a, k_final

  // 5) last-step r projection (M=16 WMMA GEMM), then the sequential scan
  gemm(XrL, WrB, RL, EBc, Hc, Hc);
  scan_kernel<<<(EBc * NHc) / SCAN_WPB, 32 * SCAN_WPB, 0, stream>>>(EW, KM, VM, NKK, BM, RL, OS);

  // 6) g gate (last step only) + epilogue + output projection
  gemm(XgL, g1T, HG, EBc, 64, Hc);
  sig_cvt_kernel<<<blks(EBc * 64), 256, 0, stream>>>(HG, TG, EBc * 64);
  gemm(TG, g2T, GL, EBc, Hc, 64);
  epilogue_kernel<<<EBc, 256, 0, stream>>>(OS, RL, KM, VM, GL, gn_w, gn_b, r_k, OG);
  gemm(OG, WoB, out, EBc, Hc, Hc);      // final (2,8,256) = d_out
}
